// BottomLevelDecoderRNN_35854386987164
// MI455X (gfx1250) — matrix operand order repaired and checked by reading, not compiled, not run on hardware
//
#include <hip/hip_runtime.h>
#include <hip/hip_bf16.h>
#include <math.h>

// ---------------------------------------------------------------------------
// Problem constants (from reference): E=16, B=128, C=512, H=1024, D=130, S=16
// N = E*B = 2048 rows, G = 3H = 3072 gate width, T = 256.
// ---------------------------------------------------------------------------
#define NROW 2048
#define CDIM 512
#define HDIM 1024
#define GDIM 3072
#define DDIM 130
#define DPAD 160          // D padded to multiple of 32 for WMMA K
#define MOUT 192          // output GEMM M padded to multiple of 64
#define SSTEPS 16
#define EDIM 16
#define BDIM 128
#define TDIM 256

typedef __attribute__((ext_vector_type(16))) __bf16 v16bf;
typedef __attribute__((ext_vector_type(8)))  __bf16 v8bf;
typedef __attribute__((ext_vector_type(8)))  float  v8f;

union BF16Frag { v16bf v; v8bf h[2]; };

// ---------------------------------------------------------------------------
// bf16 WMMA GEMM:  Out[N x M](f32) = A[N x K](bf16) * W[M x K]^T(bf16) [+ Cin]
// Block = 256 threads (8 waves). Block tile = 256 rows x 64 cols.
// Wave w owns rows [blockIdx.y*256 + w*32, +32) x cols [blockIdx.x*64, +64):
//   2 A-fragments x 4 B-tiles -> 8 WMMAs per 32-K step, B reused across rows.
// B tile (64 cols x 32 K bf16) double-buffered in LDS (one barrier / k-step,
// next tile's global_load_b128 in flight during WMMA issue).
// Weights/activations are L2-resident on MI455X (192MB L2 >> 23MB bf16
// weights), so A fragments come straight from global.
// Requires: M % 64 == 0, N % 256 == 0, K % 32 == 0.
// ---------------------------------------------------------------------------
#define LDSB_STRIDE 40    // 32 K-elems + 8 pad -> 80B/col, conflict-free b128

__global__ __launch_bounds__(256) void gemm_bf16_wmma(
    const __bf16* __restrict__ A,    // [N, K]
    const __bf16* __restrict__ W,    // [M, K]
    const float*  __restrict__ Cin,  // [N, M] or nullptr
    float* __restrict__ Out,         // [N, M]
    int M, int K)
{
  __shared__ __bf16 ldsB[2][64 * LDSB_STRIDE];

  const int tid  = threadIdx.x;
  const int lane = tid & 31;
  const int kh   = lane >> 4;       // which 16-lane half of the wave
  const int lrow = lane & 15;
  const int wave = tid >> 5;        // 0..7

  const int m0 = blockIdx.x * 64;
  const int n0 = blockIdx.y * 256 + wave * 32;

  // B staging coords: thread t loads 8 bf16 (16B) of W[(m0+bcol), k+bk ..]
  const int bcol = tid >> 2;        // 0..63
  const int bk   = (tid & 3) * 8;   // 0,8,16,24

  v8f acc[2][4];
#pragma unroll
  for (int rt = 0; rt < 2; ++rt)
#pragma unroll
    for (int mt = 0; mt < 4; ++mt) {
      if (Cin) {
        const int col = m0 + mt * 16 + lrow;
#pragma unroll
        for (int r = 0; r < 8; ++r)
          acc[rt][mt][r] = Cin[(size_t)(n0 + rt * 16 + kh * 8 + r) * M + col];
      } else {
#pragma unroll
        for (int r = 0; r < 8; ++r) acc[rt][mt][r] = 0.0f;
      }
    }

  const __bf16* arow0 = A + (size_t)(n0 + lrow) * K;
  const __bf16* arow1 = A + (size_t)(n0 + 16 + lrow) * K;
  const __bf16* wrow  = W + (size_t)(m0 + bcol) * K + bk;

  // prologue: stage tile 0
  *(v8bf*)&ldsB[0][bcol * LDSB_STRIDE + bk] = *(const v8bf*)(wrow);

  int cur = 0;
  for (int k = 0; k < K; k += 32) {
    // prefetch next B tile into registers before the barrier
    const bool have_next = (k + 32 < K);
    v8bf nextW = {};
    if (have_next) nextW = *(const v8bf*)(wrow + k + 32);

    __syncthreads();   // ldsB[cur] writes visible; prior reads of cur^1 done

    // A fragments per ISA 16-bit A-matrix layout:
    // lanes 0-15: K 0..7 then 16..23 ; lanes 16-31: K 8..15 then 24..31
    BF16Frag af0, af1;
    af0.h[0] = *(const v8bf*)(arow0 + k + kh * 8);
    af0.h[1] = *(const v8bf*)(arow0 + k + 16 + kh * 8);
    af1.h[0] = *(const v8bf*)(arow1 + k + kh * 8);
    af1.h[1] = *(const v8bf*)(arow1 + k + 16 + kh * 8);

#pragma unroll
    for (int mt = 0; mt < 4; ++mt) {
      // B fragment: lane holds col (mt*16+lrow), K kh*8..+7 then 16+kh*8..+7
      const __bf16* bp = &ldsB[cur][(mt * 16 + lrow) * LDSB_STRIDE + kh * 8];
      BF16Frag bf;
      bf.h[0] = *(const v8bf*)bp;
      bf.h[1] = *(const v8bf*)(bp + 16);
      acc[0][mt] = __builtin_amdgcn_wmma_f32_16x16x32_bf16(
          false, af0.v, false, bf.v, (short)0, acc[0][mt], false, false);
      acc[1][mt] = __builtin_amdgcn_wmma_f32_16x16x32_bf16(
          false, af1.v, false, bf.v, (short)0, acc[1][mt], false, false);
    }

    if (have_next)
      *(v8bf*)&ldsB[cur ^ 1][bcol * LDSB_STRIDE + bk] = nextW;
    cur ^= 1;
  }

  // store: C/D layout -> lane half selects row group, VGPR r selects row
#pragma unroll
  for (int rt = 0; rt < 2; ++rt)
#pragma unroll
    for (int mt = 0; mt < 4; ++mt) {
      const int col = m0 + mt * 16 + lrow;
#pragma unroll
      for (int r = 0; r < 8; ++r)
        Out[(size_t)(n0 + rt * 16 + kh * 8 + r) * M + col] = acc[rt][mt][r];
    }
}

// ---------------------------------------------------------------------------
// Elementwise helpers
// ---------------------------------------------------------------------------
__global__ void cvt_bf16_kernel(const float* __restrict__ src,
                                __bf16* __restrict__ dst, int n) {
  int i = blockIdx.x * 256 + threadIdx.x;
  if (i < n) dst[i] = (__bf16)src[i];
}

// split g1_wih [3072, 642] into W1c [3072,512] bf16
__global__ void build_w1c_kernel(const float* __restrict__ g1wih,
                                 __bf16* __restrict__ w1c) {
  int i = blockIdx.x * 256 + threadIdx.x;
  if (i < GDIM * CDIM) {
    int m = i >> 9, k = i & 511;
    w1c[i] = (__bf16)g1wih[m * (CDIM + DDIM) + k];
  }
}

// W1p [3072, 160] bf16, cols >=130 zero
__global__ void build_w1p_kernel(const float* __restrict__ g1wih,
                                 __bf16* __restrict__ w1p) {
  int i = blockIdx.x * 256 + threadIdx.x;
  if (i < GDIM * DPAD) {
    int m = i / DPAD, k = i % DPAD;
    w1p[i] = (k < DDIM) ? (__bf16)g1wih[m * (CDIM + DDIM) + CDIM + k]
                        : (__bf16)0.0f;
  }
}

// fco_w [130,1024] -> [192,1024] bf16 zero-padded rows
__global__ void build_fco_kernel(const float* __restrict__ fcow,
                                 __bf16* __restrict__ wo) {
  int i = blockIdx.x * 256 + threadIdx.x;
  if (i < MOUT * HDIM) {
    int m = i >> 10, k = i & 1023;
    wo[i] = (m < DDIM) ? (__bf16)fcow[m * HDIM + k] : (__bf16)0.0f;
  }
}

// teacher-forced prev for all steps: prev[s][e*B+b][d] (d padded to 160)
__global__ void build_prev_kernel(const float* __restrict__ target,
                                  __bf16* __restrict__ prevb) {
  int i = blockIdx.x * 256 + threadIdx.x;
  if (i < SSTEPS * NROW * DPAD) {
    int d = i % DPAD;
    int r = (i / DPAD) % NROW;
    int s = i / (NROW * DPAD);
    int e = r >> 7, b = r & 127;
    int t = e * SSTEPS + s;
    float v = 0.0f;
    if (d < DDIM && t > 0)
      v = target[((size_t)b * TDIM + (t - 1)) * DDIM + d];
    prevb[i] = (__bf16)v;
  }
}

// h init: t0raw [2048,2048] raw GEMM -> h1/h2 = tanh(t0 + b)
__global__ void init_h_kernel(const float* __restrict__ t0raw,
                              const float* __restrict__ fib,
                              float* __restrict__ h1f, __bf16* __restrict__ h1b,
                              float* __restrict__ h2f, __bf16* __restrict__ h2b) {
  int i = blockIdx.x * 256 + threadIdx.x;
  if (i < NROW * HDIM) {
    int n = i >> 10, j = i & 1023;
    float a = tanhf(t0raw[(size_t)n * (2 * HDIM) + j] + fib[j]);
    float b = tanhf(t0raw[(size_t)n * (2 * HDIM) + HDIM + j] + fib[HDIM + j]);
    h1f[i] = a; h1b[i] = (__bf16)a;
    h2f[i] = b; h2b[i] = (__bf16)b;
  }
}

__device__ __forceinline__ float sigmoidf_(float x) {
  return 1.0f / (1.0f + __expf(-x));
}

// GRU gate combine: h_new = (1-z)*n + z*h  (in-place safe, per element)
__global__ void gru_gate_kernel(const float* __restrict__ gi,
                                const float* __restrict__ gh,
                                const float* __restrict__ bih,
                                const float* __restrict__ bhh,
                                const float* __restrict__ hin,
                                float* __restrict__ hf,
                                __bf16* __restrict__ hb) {
  int i = blockIdx.x * 256 + threadIdx.x;
  if (i < NROW * HDIM) {
    int n = i >> 10, j = i & 1023;
    size_t base = (size_t)n * GDIM;
    float r  = sigmoidf_(gi[base + j] + bih[j] + gh[base + j] + bhh[j]);
    float z  = sigmoidf_(gi[base + HDIM + j] + bih[HDIM + j] +
                         gh[base + HDIM + j] + bhh[HDIM + j]);
    float nn = tanhf(gi[base + 2 * HDIM + j] + bih[2 * HDIM + j] +
                     r * (gh[base + 2 * HDIM + j] + bhh[2 * HDIM + j]));
    float h  = (1.0f - z) * nn + z * hin[i];
    hf[i] = h; hb[i] = (__bf16)h;
  }
}

// scatter y[2048,192] -> out[b, e*S+s, d] (+bias), d<130
__global__ void scatter_out_kernel(const float* __restrict__ yraw,
                                   const float* __restrict__ fcob,
                                   float* __restrict__ out, int s) {
  int i = blockIdx.x * 256 + threadIdx.x;
  if (i < NROW * DDIM) {
    int r = i / DDIM, d = i % DDIM;
    int e = r >> 7, b = r & 127;
    int t = e * SSTEPS + s;
    out[((size_t)b * TDIM + t) * DDIM + d] = yraw[(size_t)r * MOUT + d] + fcob[d];
  }
}

// ---------------------------------------------------------------------------
// Host-side launch
// ---------------------------------------------------------------------------
static inline char* ws_take(char* base, size_t& off, size_t bytes) {
  off = (off + 255) & ~(size_t)255;
  char* p = base + off;
  off += bytes;
  return p;
}

extern "C" void kernel_launch(void* const* d_in, const int* in_sizes, int n_in,
                              void* d_out, int out_size, void* d_ws, size_t ws_size,
                              hipStream_t stream) {
  const float* c      = (const float*)d_in[0];
  const float* target = (const float*)d_in[1];
  const float* fiw    = (const float*)d_in[4];
  const float* fib    = (const float*)d_in[5];
  const float* g1wih  = (const float*)d_in[6];
  const float* g1whh  = (const float*)d_in[7];
  const float* g1bih  = (const float*)d_in[8];
  const float* g1bhh  = (const float*)d_in[9];
  const float* g2wih  = (const float*)d_in[10];
  const float* g2whh  = (const float*)d_in[11];
  const float* g2bih  = (const float*)d_in[12];
  const float* g2bhh  = (const float*)d_in[13];
  const float* fcow   = (const float*)d_in[14];
  const float* fcob   = (const float*)d_in[15];
  float* out = (float*)d_out;

  char* ws = (char*)d_ws;
  size_t off = 0;
  // bf16 buffers
  __bf16* cbf    = (__bf16*)ws_take(ws, off, (size_t)NROW * CDIM * 2);
  __bf16* fiwb   = (__bf16*)ws_take(ws, off, (size_t)(2 * HDIM) * CDIM * 2);
  __bf16* w1c    = (__bf16*)ws_take(ws, off, (size_t)GDIM * CDIM * 2);
  __bf16* w1p    = (__bf16*)ws_take(ws, off, (size_t)GDIM * DPAD * 2);
  __bf16* whh1b  = (__bf16*)ws_take(ws, off, (size_t)GDIM * HDIM * 2);
  __bf16* wih2b  = (__bf16*)ws_take(ws, off, (size_t)GDIM * HDIM * 2);
  __bf16* whh2b  = (__bf16*)ws_take(ws, off, (size_t)GDIM * HDIM * 2);
  __bf16* fcowb  = (__bf16*)ws_take(ws, off, (size_t)MOUT * HDIM * 2);
  __bf16* prevb  = (__bf16*)ws_take(ws, off, (size_t)SSTEPS * NROW * DPAD * 2);
  __bf16* h1b    = (__bf16*)ws_take(ws, off, (size_t)NROW * HDIM * 2);
  __bf16* h2b    = (__bf16*)ws_take(ws, off, (size_t)NROW * HDIM * 2);
  // f32 buffers
  float* h1f     = (float*)ws_take(ws, off, (size_t)NROW * HDIM * 4);
  float* h2f     = (float*)ws_take(ws, off, (size_t)NROW * HDIM * 4);
  float* base1   = (float*)ws_take(ws, off, (size_t)NROW * GDIM * 4);
  float* gi      = (float*)ws_take(ws, off, (size_t)NROW * GDIM * 4);   // also t0raw
  float* gh      = (float*)ws_take(ws, off, (size_t)NROW * GDIM * 4);
  float* yraw    = (float*)ws_take(ws, off, (size_t)NROW * MOUT * 4);
  float* t0raw   = gi;  // alias: init-only use, before first gi GEMM

  const dim3 blk(256);
  auto eb = [](int n) { return dim3((unsigned)((n + 255) / 256)); };
  auto gemm = [&](const __bf16* A, const __bf16* W, const float* Cin,
                  float* O, int M, int K) {
    dim3 grid((unsigned)(M / 64), (unsigned)(NROW / 256));
    gemm_bf16_wmma<<<grid, blk, 0, stream>>>(A, W, Cin, O, M, K);
  };

  // ---- one-time prep (recomputed every call; fully deterministic) ----
  cvt_bf16_kernel<<<eb(NROW * CDIM), blk, 0, stream>>>(c, cbf, NROW * CDIM);
  cvt_bf16_kernel<<<eb(2 * HDIM * CDIM), blk, 0, stream>>>(fiw, fiwb, 2 * HDIM * CDIM);
  cvt_bf16_kernel<<<eb(GDIM * HDIM), blk, 0, stream>>>(g1whh, whh1b, GDIM * HDIM);
  cvt_bf16_kernel<<<eb(GDIM * HDIM), blk, 0, stream>>>(g2wih, wih2b, GDIM * HDIM);
  cvt_bf16_kernel<<<eb(GDIM * HDIM), blk, 0, stream>>>(g2whh, whh2b, GDIM * HDIM);
  build_w1c_kernel<<<eb(GDIM * CDIM), blk, 0, stream>>>(g1wih, w1c);
  build_w1p_kernel<<<eb(GDIM * DPAD), blk, 0, stream>>>(g1wih, w1p);
  build_fco_kernel<<<eb(MOUT * HDIM), blk, 0, stream>>>(fcow, fcowb);
  build_prev_kernel<<<eb(SSTEPS * NROW * DPAD), blk, 0, stream>>>(target, prevb);

  // h1/h2 init: t0 = tanh(c @ fc_init_w^T + b)
  gemm(cbf, fiwb, nullptr, t0raw, 2 * HDIM, CDIM);
  init_h_kernel<<<eb(NROW * HDIM), blk, 0, stream>>>(t0raw, fib, h1f, h1b, h2f, h2b);

  // base1 = c @ W1c^T  (step-invariant half of GRU1 input GEMM)
  gemm(cbf, w1c, nullptr, base1, GDIM, CDIM);

  // ---- sequential scan over S = 16 steps ----
  for (int s = 0; s < SSTEPS; ++s) {
    const __bf16* prev_s = prevb + (size_t)s * NROW * DPAD;
    // GRU1: gi = base1 + prev @ W1p^T ; gh = h1 @ whh1^T
    gemm(prev_s, w1p, base1, gi, GDIM, DPAD);
    gemm(h1b, whh1b, nullptr, gh, GDIM, HDIM);
    gru_gate_kernel<<<eb(NROW * HDIM), blk, 0, stream>>>(gi, gh, g1bih, g1bhh,
                                                         h1f, h1f, h1b);
    // GRU2: gi = h1 @ wih2^T ; gh = h2 @ whh2^T
    gemm(h1b, wih2b, nullptr, gi, GDIM, HDIM);
    gemm(h2b, whh2b, nullptr, gh, GDIM, HDIM);
    gru_gate_kernel<<<eb(NROW * HDIM), blk, 0, stream>>>(gi, gh, g2bih, g2bhh,
                                                         h2f, h2f, h2b);
    // y = h2 @ fco_w^T + b  -> scatter to [B, T, D]
    gemm(h2b, fcowb, nullptr, yraw, MOUT, HDIM);
    scatter_out_kernel<<<eb(NROW * DDIM), blk, 0, stream>>>(yraw, fcob, out, s);
  }
}
